// RGNN_49237505081950
// MI455X (gfx1250) — matrix operand order repaired and checked by reading, not compiled, not run on hardware
//
#include <hip/hip_runtime.h>

typedef __attribute__((ext_vector_type(16))) _Float16 v16h;
typedef __attribute__((ext_vector_type(8)))  _Float16 v8h;
typedef __attribute__((ext_vector_type(8)))  float    v8f;

// ---------------------------------------------------------------------------
// helpers
// ---------------------------------------------------------------------------
__device__ __forceinline__ void atomicMaxF(float* addr, float v) {
  // sign-split trick: int compare is monotonic for >=0 floats,
  // uint compare is reverse-monotonic for <0 floats.
  if (v >= 0.0f) atomicMax((int*)addr, __float_as_int(v));
  else           atomicMin((unsigned int*)addr, __float_as_uint(v));
}

__global__ void k_fill(float* __restrict__ p, float v, int n) {
  int i = blockIdx.x * blockDim.x + threadIdx.x;
  if (i < n) p[i] = v;
}

// ---------------------------------------------------------------------------
// Pack W[128,Nreal] (fp32, row-major) into per-lane WMMA-B fragment layout
// (f16). Element (nt, kbi, lane, i) at idx = nt*2048 + kbi*512 + lane*16 + i
// holds B[K = kbi*32 + (lane>=16)*16 + i, col = nt*16 + (lane&15)].
// Out-of-range columns are zero-filled so the GEMM needs no B guard.
// ---------------------------------------------------------------------------
__global__ void k_packB(const float* __restrict__ W, _Float16* __restrict__ Bp,
                        int Nreal, int total) {
  int idx = blockIdx.x * blockDim.x + threadIdx.x;
  if (idx >= total) return;
  int i    = idx & 15;
  int lane = (idx >> 4) & 31;
  int kbi  = (idx >> 9) & 3;
  int nt   = idx >> 11;
  int K    = kbi * 32 + ((lane >> 4) << 4) + i;
  int col  = nt * 16 + (lane & 15);
  Bp[idx] = (col < Nreal) ? (_Float16)W[(size_t)K * Nreal + col] : (_Float16)0.0f;
}

// ---------------------------------------------------------------------------
// GEMM: out[M,Nreal] = X[M,128] * B (+ bias), f16 WMMA, f32 accum
// block = 256 threads (8 waves). blockIdx.x = M-tile, blockIdx.y*8+wave = N-tile
// B comes pre-packed in fragment layout: two b128 loads per K-step per lane.
// ---------------------------------------------------------------------------
__global__ void k_gemm_f16wmma(const float* __restrict__ X,
                               const _Float16* __restrict__ Bp,
                               const float* __restrict__ bias,
                               float* __restrict__ out,
                               int M, int Nreal) {
  __shared__ __align__(16) _Float16 As[16 * 128];
  const int tid = threadIdx.x;
  const int m0  = blockIdx.x * 16;

  // stage 16x128 A tile as f16 into LDS (coalesced)
  #pragma unroll
  for (int i = 0; i < 8; ++i) {
    int idx = tid + 256 * i;
    int r = idx >> 7, c = idx & 127;
    int gr = m0 + r;
    As[idx] = (gr < M) ? (_Float16)X[(size_t)gr * 128 + c] : (_Float16)0.0f;
  }
  __syncthreads();

  const int wave = tid >> 5;
  const int lane = tid & 31;
  const int ntiles = (Nreal + 15) >> 4;
  const int nt = blockIdx.y * 8 + wave;
  if (nt >= ntiles) return;
  const int col0 = nt << 4;

  // A fragment addressing (contiguous K order, matching packed B):
  //   lane<16: M=lane, halves 0..15 -> K=0..15
  //   lane>=16: same M, K offset by 16
  const int rowA = lane & 15;
  const int koff = (lane >> 4) << 4;   // 0 or 16
  const _Float16* bp = Bp + (size_t)nt * 2048 + (size_t)lane * 16;

  v8f acc = {};
  #pragma unroll
  for (int kbi = 0; kbi < 4; ++kbi) {
    const int kb = kbi * 32;
    v16h a, b;
    v8h a0 = *(const v8h*)&As[rowA * 128 + kb + koff];
    v8h a1 = *(const v8h*)&As[rowA * 128 + kb + koff + 8];
    v8h b0 = *(const v8h*)(bp + kbi * 512);
    v8h b1 = *(const v8h*)(bp + kbi * 512 + 8);
    #pragma unroll
    for (int i = 0; i < 8; ++i) {
      a[i] = a0[i]; a[i + 8] = a1[i];
      b[i] = b0[i]; b[i + 8] = b1[i];
    }
    acc = __builtin_amdgcn_wmma_f32_16x16x32_f16(false, a, false, b,
                                                 (short)0, acc, false, false);
  }

  const int colO = col0 + (lane & 15);
  if (colO < Nreal) {
    float bv = bias ? bias[colO] : 0.0f;
    int rbase = m0 + ((lane >> 4) << 3);  // C layout: VGPR r -> M = r + 8*(lane>=16)
    #pragma unroll
    for (int r = 0; r < 8; ++r) {
      int gr = rbase + r;
      if (gr < M) out[(size_t)gr * Nreal + colO] = acc[r] + bv;
    }
  }
}

// ---------------------------------------------------------------------------
// attention scalars: a[n,h] = sum_c h[n, h*32+c] * att[h*32+c]
// block = 128 threads (4 waves, one per head), grid = N
// ---------------------------------------------------------------------------
__global__ void k_att(const float* __restrict__ h, const float* __restrict__ att,
                      float* __restrict__ a) {
  const int n = blockIdx.x;
  const int j = threadIdx.x;          // 0..127, head = j>>5
  float v = h[(size_t)n * 128 + j] * att[j];
  #pragma unroll
  for (int off = 16; off > 0; off >>= 1)
    v += __shfl_xor(v, off, 32);
  if ((j & 31) == 0) a[n * 4 + (j >> 5)] = v;
}

// ---------------------------------------------------------------------------
// edge pass 1: e = leaky_relu(a_s[src]+a_d[dst], 0.2); segment max via atomics
// ---------------------------------------------------------------------------
__global__ void k_edge1(const int* __restrict__ src, const int* __restrict__ dst,
                        const float* __restrict__ as, const float* __restrict__ ad,
                        float* __restrict__ ebuf, float* __restrict__ m, int EH) {
  int i = blockIdx.x * blockDim.x + threadIdx.x;
  if (i >= EH) return;
  int e = i >> 2, hh = i & 3;
  int s = src[e], d = dst[e];
  float v = as[s * 4 + hh] + ad[d * 4 + hh];
  v = (v > 0.0f) ? v : 0.2f * v;
  ebuf[i] = v;
  atomicMaxF(&m[d * 4 + hh], v);
}

// edge pass 2: w = exp(e - m[dst]); z[dst] += w
__global__ void k_edge2(const int* __restrict__ dst, float* __restrict__ ebuf,
                        const float* __restrict__ m, float* __restrict__ z, int EH) {
  int i = blockIdx.x * blockDim.x + threadIdx.x;
  if (i >= EH) return;
  int e = i >> 2, hh = i & 3;
  int d = dst[e];
  float w = __expf(ebuf[i] - m[d * 4 + hh]);
  ebuf[i] = w;
  atomicAdd(&z[d * 4 + hh], w);
}

// edge pass 3: agg[dst, j] += h_s[src, j] * alpha[head(j)]
// one wave per edge; lane = channel within head chunk; 8 edges per block
__global__ void k_edge3(const int* __restrict__ src, const int* __restrict__ dst,
                        const float* __restrict__ ebuf, const float* __restrict__ z,
                        const float* __restrict__ hs, float* __restrict__ agg, int E) {
  int e = blockIdx.x * 8 + (threadIdx.x >> 5);
  int lane = threadIdx.x & 31;
  if (e >= E) return;
  int s = src[e], d = dst[e];
  float al = 0.0f;
  if (lane < 4) al = ebuf[e * 4 + lane] / (z[d * 4 + lane] + 1e-16f);
  #pragma unroll
  for (int k = 0; k < 4; ++k) {
    float ak = __shfl(al, k, 32);
    int j = k * 32 + lane;                 // head = k
    atomicAdd(&agg[(size_t)d * 128 + j], hs[(size_t)s * 128 + j] * ak);
  }
}

// finalize: x = act(agg + b1 + b2?) , act = leaky_relu(0.01) or identity
__global__ void k_finalize(float* __restrict__ x, const float* __restrict__ b1,
                           const float* __restrict__ b2, int n, int act) {
  int i = blockIdx.x * blockDim.x + threadIdx.x;
  if (i >= n) return;
  int j = i & 127;
  float v = x[i] + b1[j] + (b2 ? b2[j] : 0.0f);
  if (act && v < 0.0f) v *= 0.01f;
  x[i] = v;
}

// ---------------------------------------------------------------------------
// host-side GEMM wrapper: pack weights, then WMMA GEMM
// ---------------------------------------------------------------------------
static void gemm(const float* X, const float* W, const float* bias, float* out,
                 int M, int Nreal, _Float16* Bp, hipStream_t st) {
  int ntiles = (Nreal + 15) / 16;
  int total  = ntiles * 2048;
  k_packB<<<(total + 255) / 256, 256, 0, st>>>(W, Bp, Nreal, total);
  dim3 g((M + 15) / 16, (ntiles + 7) / 8);
  k_gemm_f16wmma<<<g, 256, 0, st>>>(X, Bp, bias, out, M, Nreal);
}

// host-side relation driver (all launches on stream)
static void run_relation(const float* xs, const float* xd, int same,
                         const float* W, const float* att_s, const float* att_d,
                         const int* ei, int E, int Nn,
                         float* agg, float* hA, float* hB,
                         float* a_s, float* a_d, float* m, float* z, float* ebuf,
                         _Float16* Bp, hipStream_t st) {
  gemm(xs, W, nullptr, hA, Nn, 128, Bp, st);
  k_att<<<Nn, 128, 0, st>>>(hA, att_s, a_s);
  const float* hd = hA;
  if (!same) {
    gemm(xd, W, nullptr, hB, Nn, 128, Bp, st);
    hd = hB;
  }
  k_att<<<Nn, 128, 0, st>>>(hd, att_d, a_d);

  int NH4 = Nn * 4;
  k_fill<<<(NH4 + 255) / 256, 256, 0, st>>>(m, -1e30f, NH4);
  k_fill<<<(NH4 + 255) / 256, 256, 0, st>>>(z, 0.0f, NH4);

  const int* src = ei;
  const int* dst = ei + E;
  int EH = E * 4;
  k_edge1<<<(EH + 255) / 256, 256, 0, st>>>(src, dst, a_s, a_d, ebuf, m, EH);
  k_edge2<<<(EH + 255) / 256, 256, 0, st>>>(dst, ebuf, m, z, EH);
  k_edge3<<<(E + 7) / 8, 256, 0, st>>>(src, dst, ebuf, z, hA, agg, E);
}

extern "C" void kernel_launch(void* const* d_in, const int* in_sizes, int n_in,
                              void* d_out, int out_size, void* d_ws, size_t ws_size,
                              hipStream_t stream) {
  const float* xp    = (const float*)d_in[0];
  const float* xa    = (const float*)d_in[1];
  const int*   ei_c  = (const int*)d_in[2];
  const int*   ei_w  = (const int*)d_in[3];
  const int*   ei_wb = (const int*)d_in[4];
  const int Nn  = in_sizes[0] / 128;
  const int E   = in_sizes[2] / 2;
  const int OUT = in_sizes[30];
  auto P = [&](int i) { return (const float*)d_in[i]; };
  // param layout: 5:W0_c 6:as0_c 7:ad0_c 8:b0_c 9:W0_w 10:as0_w 11:ad0_w 12:b0_w
  //               13:W0_wb 14:as0_wb 15:ad0_wb 16:b0_wb ; layer1 at 17..28 ; 29:lin_W 30:lin_b

  float* ws = (float*)d_ws;
  float* hA    = ws; ws += (size_t)Nn * 128;
  float* hB    = ws; ws += (size_t)Nn * 128;
  float* aggP0 = ws; ws += (size_t)Nn * 128;
  float* aggA0 = ws; ws += (size_t)Nn * 128;
  float* aggP1 = ws; ws += (size_t)Nn * 128;
  float* a_s   = ws; ws += (size_t)Nn * 4;
  float* a_d   = ws; ws += (size_t)Nn * 4;
  float* mbuf  = ws; ws += (size_t)Nn * 4;
  float* zbuf  = ws; ws += (size_t)Nn * 4;
  float* ebuf  = ws; ws += (size_t)E * 4;
  _Float16* Bp = (_Float16*)ws; ws += 32768;   // up to 64K halfs (>= 22 tiles * 2048)

  const int nf = Nn * 128;
  const int gf = (nf + 255) / 256;

  // ----- layer 0 -----
  k_fill<<<gf, 256, 0, stream>>>(aggP0, 0.0f, nf);
  k_fill<<<gf, 256, 0, stream>>>(aggA0, 0.0f, nf);
  // cites: paper->paper (src==dst features)
  run_relation(xp, xp, 1, P(5),  P(6),  P(7),  ei_c,  E, Nn, aggP0,
               hA, hB, a_s, a_d, mbuf, zbuf, ebuf, Bp, stream);
  // writes: author->paper
  run_relation(xa, xp, 0, P(9),  P(10), P(11), ei_w,  E, Nn, aggP0,
               hA, hB, a_s, a_d, mbuf, zbuf, ebuf, Bp, stream);
  // written_by: paper->author
  run_relation(xp, xa, 0, P(13), P(14), P(15), ei_wb, E, Nn, aggA0,
               hA, hB, a_s, a_d, mbuf, zbuf, ebuf, Bp, stream);
  k_finalize<<<gf, 256, 0, stream>>>(aggP0, P(8),  P(12),   nf, 1);
  k_finalize<<<gf, 256, 0, stream>>>(aggA0, P(16), nullptr, nf, 1);

  // ----- layer 1 (author output is never used by the head -> skip wb) -----
  k_fill<<<gf, 256, 0, stream>>>(aggP1, 0.0f, nf);
  run_relation(aggP0, aggP0, 1, P(17), P(18), P(19), ei_c, E, Nn, aggP1,
               hA, hB, a_s, a_d, mbuf, zbuf, ebuf, Bp, stream);
  run_relation(aggA0, aggP0, 0, P(21), P(22), P(23), ei_w, E, Nn, aggP1,
               hA, hB, a_s, a_d, mbuf, zbuf, ebuf, Bp, stream);
  k_finalize<<<gf, 256, 0, stream>>>(aggP1, P(20), P(24), nf, 0);

  // ----- linear head: out = xp2 @ lin_W + lin_b -----
  gemm(aggP1, P(29), P(30), (float*)d_out, Nn, OUT, Bp, stream);
}
